// CausalSelfAttention_61641370632465
// MI455X (gfx1250) — matrix operand order repaired and checked by reading, not compiled, not run on hardware
//
#include <hip/hip_runtime.h>
#include <hip/hip_bf16.h>
#include <math.h>

// ---------------------------------------------------------------------------
// CDNA5 (gfx1250) causal GQA attention with RoPE.
// bf16 WMMA for all matmuls; TDM (tensor_load_to_lds) double-buffered LDS GEMM.
// ---------------------------------------------------------------------------

typedef __attribute__((ext_vector_type(16))) __bf16 v16bf;
typedef __attribute__((ext_vector_type(8)))  __bf16 v8bf;
typedef __attribute__((ext_vector_type(8)))  float  v8f;
typedef __attribute__((ext_vector_type(4)))  unsigned v4u;
typedef __attribute__((ext_vector_type(8)))  int      v8i;
typedef __attribute__((ext_vector_type(4)))  int      v4i;

union V16 { v16bf v; v8bf h[2]; };

constexpr int Tn  = 2048;   // sequence length
constexpr int Cn  = 2048;   // model dim
constexpr int Hn  = 16;     // query heads
constexpr int KHn = 4;      // kv heads
constexpr int HDn = 128;    // head dim

static __device__ __forceinline__ v8f wmma_bf16(const V16& a, const V16& b, v8f c) {
  return __builtin_amdgcn_wmma_f32_16x16x32_bf16(false, a.v, false, b.v,
                                                 (short)0, c, false, false);
}

// ---------------------- TDM 2D tile load: global -> LDS --------------------
// Builds a Tensor DMA Descriptor (D#) per CDNA5 ISA ch.8 and issues
// tensor_load_to_lds. tile_w elements per row, tile_h rows, bf16 (2B) data.
static __device__ __forceinline__ void tdm_load_2d(
    unsigned lds_off, const void* gaddr,
    unsigned tile_w, unsigned tile_h,
    unsigned tensor_w, unsigned tensor_h, unsigned row_stride /*elems*/) {
  unsigned long long ga = (unsigned long long)(size_t)gaddr;
  v4u g0;
  g0.x = 1u;                                        // count=1, user descriptor
  g0.y = lds_off;                                   // lds_addr (bytes)
  g0.z = (unsigned)(ga & 0xFFFFFFFFu);              // global_addr[31:0]
  g0.w = (unsigned)((ga >> 32) & 0x01FFFFFFu)       // global_addr[56:32]
       | (2u << 30);                                // type = 2 ("image")
  v8i g1;
  g1[0] = (int)(1u << 16);                          // wg_mask=0, data_size=1 (2B)
  g1[1] = (int)((tensor_w & 0xFFFFu) << 16);        // tensor_dim0[15:0] @ bits 63:48
  g1[2] = (int)(((tensor_w >> 16) & 0xFFFFu) | ((tensor_h & 0xFFFFu) << 16));
  g1[3] = (int)(((tensor_h >> 16) & 0xFFFFu) | ((tile_w & 0xFFFFu) << 16));
  g1[4] = (int)(tile_h & 0xFFFFu);                  // tile_dim1; tile_dim2=0 (2D)
  g1[5] = (int)row_stride;                          // tensor_dim0_stride[31:0]
  g1[6] = 0;                                        // stride hi, dim1_stride lo
  g1[7] = 0;
  v4i g2 = {0, 0, 0, 0};
  v4i g3 = {0, 0, 0, 0};
#if __has_include(<hip/amd_detail/amd_gfx1250_TDM.h>)
  v8i g4 = {0, 0, 0, 0, 0, 0, 0, 0};
  __builtin_amdgcn_tensor_load_to_lds(g0, g1, g2, g3, g4, 0);
#else
  __builtin_amdgcn_tensor_load_to_lds(g0, g1, g2, g3, 0);
#endif
}

// -------------------------- fp32 -> bf16 convert ---------------------------
__global__ void cvt_bf16_kernel(const float* __restrict__ src,
                                __bf16* __restrict__ dst, size_t n) {
  size_t i = (size_t)blockIdx.x * blockDim.x + threadIdx.x;
  if (i < n) dst[i] = (__bf16)src[i];
}

// ---------------- W [K][N] fp32  ->  Wt [N][K] bf16 (transpose) ------------
__global__ void transpose_bf16_kernel(const float* __restrict__ W,
                                      __bf16* __restrict__ Wt, int K, int N) {
  size_t idx = (size_t)blockIdx.x * blockDim.x + threadIdx.x;
  if (idx >= (size_t)K * N) return;
  int n = (int)(idx / K);
  int k = (int)(idx % K);
  Wt[idx] = (__bf16)W[(size_t)k * N + n];
}

// ------------------ TDM double-buffered LDS bf16 WMMA GEMM -----------------
// C[M][N] (f32) = A[M][K] (bf16, row-major) * Bt[N][K]^T (bf16)
// Block: 512 threads = 16 waves -> 128x128 output tile, 32-deep K slabs.
// Wave 0 issues TDM DMAs for slab k+1 while all waves run WMMA on slab k.
__global__ __launch_bounds__(512, 1) void gemm_bf16_tdm_kernel(
    const __bf16* __restrict__ A, const __bf16* __restrict__ Bt,
    float* __restrict__ C, int M, int N, int K) {
  __shared__ __align__(128) __bf16 sA[2][128 * 32];   // 16 KB
  __shared__ __align__(128) __bf16 sB[2][128 * 32];   // 16 KB

  int lane = threadIdx.x & 31;
  int wv   = threadIdx.x >> 5;        // 0..15
  int wr   = wv >> 2, wc = wv & 3;    // 4x4 wave grid, 32x32 each
  int ntb  = N >> 7;
  int bm   = blockIdx.x / ntb, bn = blockIdx.x % ntb;
  int lane15 = lane & 15, hi = lane >> 4;

  const __bf16* Ablk = A  + (size_t)(bm * 128) * K;
  const __bf16* Bblk = Bt + (size_t)(bn * 128) * K;

  // prologue: DMA first slabs into buffer 0
  if (wv == 0) {
    tdm_load_2d((unsigned)(size_t)&sA[0][0], Ablk, 32, 128, (unsigned)K, (unsigned)M, (unsigned)K);
    tdm_load_2d((unsigned)(size_t)&sB[0][0], Bblk, 32, 128, (unsigned)K, (unsigned)N, (unsigned)K);
    __builtin_amdgcn_s_wait_tensorcnt(0);
  }
  __syncthreads();

  v8f c00{}, c01{}, c10{}, c11{};
  int rowA = wr * 32 + lane15;
  int rowB = wc * 32 + lane15;
  int ka = hi * 8, ks = hi * 8;

  for (int k0 = 0; k0 < K; k0 += 32) {
    int buf = (k0 >> 5) & 1;
    bool more = (k0 + 32) < K;
    if (wv == 0 && more) {            // async DMA of next slab into other buffer
      tdm_load_2d((unsigned)(size_t)&sA[buf ^ 1][0], Ablk + (k0 + 32), 32, 128,
                  (unsigned)K, (unsigned)M, (unsigned)K);
      tdm_load_2d((unsigned)(size_t)&sB[buf ^ 1][0], Bblk + (k0 + 32), 32, 128,
                  (unsigned)K, (unsigned)N, (unsigned)K);
    }
    // fragments from LDS (layouts match wave32 WMMA A/B striping)
    V16 a0, a1, b0, b1;
    const __bf16* pa0 = &sA[buf][rowA * 32 + ka];
    const __bf16* pa1 = &sA[buf][(rowA + 16) * 32 + ka];
    a0.h[0] = *(const v8bf*)pa0;  a0.h[1] = *(const v8bf*)(pa0 + 16);
    a1.h[0] = *(const v8bf*)pa1;  a1.h[1] = *(const v8bf*)(pa1 + 16);
    const __bf16* pb0 = &sB[buf][rowB * 32 + ks];
    const __bf16* pb1 = &sB[buf][(rowB + 16) * 32 + ks];
    b0.h[0] = *(const v8bf*)pb0;  b0.h[1] = *(const v8bf*)(pb0 + 16);
    b1.h[0] = *(const v8bf*)pb1;  b1.h[1] = *(const v8bf*)(pb1 + 16);

    c00 = wmma_bf16(a0, b0, c00);
    c01 = wmma_bf16(a0, b1, c01);
    c10 = wmma_bf16(a1, b0, c10);
    c11 = wmma_bf16(a1, b1, c11);

    if (wv == 0 && more) __builtin_amdgcn_s_wait_tensorcnt(0);
    __syncthreads();                  // publish next buffer / protect current
  }

  // C/D layout: VGPR j -> row j + (lane>=16 ? 8 : 0), col = lane15
  int mbase = bm * 128 + wr * 32 + hi * 8;
  int nbase = bn * 128 + wc * 32 + lane15;
  for (int j = 0; j < 8; ++j) {
    C[(size_t)(mbase + j)      * N + nbase]      = c00[j];
    C[(size_t)(mbase + j)      * N + nbase + 16] = c01[j];
    C[(size_t)(mbase + 16 + j) * N + nbase]      = c10[j];
    C[(size_t)(mbase + 16 + j) * N + nbase + 16] = c11[j];
  }
}

// ------------------------------- RoPE --------------------------------------
__global__ void rope_kernel(const float* __restrict__ Qraw,
                            const float* __restrict__ Kraw,
                            const int* __restrict__ pos_ids,
                            __bf16* __restrict__ Qb, __bf16* __restrict__ Kb) {
  int t = blockIdx.x;
  int d = threadIdx.x;          // 0..127
  int hy = blockIdx.y;          // 0..H+KH-1
  float pos = (float)pos_ids[t];
  int i = d & 63;
  float freq = __powf(10000.0f, -((float)(2 * i)) * (1.0f / 128.0f));
  float ang = pos * freq;
  float cs = __cosf(ang), sn = __sinf(ang);
  if (hy < Hn) {
    const float* base = Qraw + (size_t)t * (Hn * HDn) + hy * HDn;
    float a   = base[d];
    float rot = (d < 64) ? -base[d + 64] : base[d - 64];
    Qb[((size_t)hy * Tn + t) * HDn + d] = (__bf16)(a * cs + rot * sn);
  } else {
    int kh = hy - Hn;
    const float* base = Kraw + (size_t)t * (KHn * HDn) + kh * HDn;
    float a   = base[d];
    float rot = (d < 64) ? -base[d + 64] : base[d - 64];
    Kb[((size_t)kh * Tn + t) * HDn + d] = (__bf16)(a * cs + rot * sn);
  }
}

// Vraw [T][KH*HD] f32 -> Vt [KH*HD][T] bf16 (transpose, B-frag friendly)
__global__ void prep_v_kernel(const float* __restrict__ Vraw,
                              __bf16* __restrict__ Vt) {
  int t   = blockIdx.x * blockDim.x + threadIdx.x;
  int col = blockIdx.y;  // 0..KH*HD-1
  Vt[(size_t)col * Tn + t] = (__bf16)Vraw[(size_t)t * (KHn * HDn) + col];
}

// --------------------- flash attention (one wave = 16 q rows) --------------
__global__ __launch_bounds__(128, 1) void attention_kernel(
    const __bf16* __restrict__ Qb,   // [H][T][HD]
    const __bf16* __restrict__ Kb,   // [KH][T][HD]
    const __bf16* __restrict__ Vt,   // [KH][HD][T]
    __bf16* __restrict__ Yb) {       // [T][H*HD]
  __shared__ __bf16 sP[4][16 * 32];

  int lane = threadIdx.x & 31;
  int wv   = threadIdx.x >> 5;
  int w    = blockIdx.x * 4 + wv;
  const int qtiles = Tn / 16;                 // 128
  int h  = w / qtiles;
  int qt = w % qtiles;
  int kh = h / (Hn / KHn);
  int q0 = qt * 16;
  int lane15 = lane & 15, hi = lane >> 4;
  const float scale = 0.08838834764831845f;   // 1/sqrt(128)

  // Preload 16x128 Q tile as four A fragments (K chunks of 32).
  V16 aQ[4];
  const __bf16* qrow = Qb + ((size_t)h * Tn + q0 + lane15) * HDn;
  for (int kc = 0; kc < 4; ++kc) {
    int ka = kc * 32 + hi * 8;
    aQ[kc].h[0] = *(const v8bf*)(qrow + ka);
    aQ[kc].h[1] = *(const v8bf*)(qrow + ka + 16);
  }

  float mrow[8], lrow[8], f[8];
  v8f O[8];
  for (int j = 0; j < 8; ++j) { mrow[j] = -3.0e38f; lrow[j] = 0.f; O[j] = v8f{}; }

  int nkb = (q0 + 47) >> 5;                   // key blocks of 32, up to diagonal
  for (int kbI = 0; kbI < nkb; ++kbI) {
    int s0 = kbI * 32;
    // ---- S = Q * K^T (two 16x16 tiles over 32 keys) ----
    v8f S0{}, S1{};
    const __bf16* kptr0 = Kb + ((size_t)kh * Tn + s0 + lane15) * HDn;
    const __bf16* kptr1 = kptr0 + (size_t)16 * HDn;
    for (int kc = 0; kc < 4; ++kc) {
      int ks = kc * 32 + hi * 8;
      V16 b0, b1;
      b0.h[0] = *(const v8bf*)(kptr0 + ks);
      b0.h[1] = *(const v8bf*)(kptr0 + ks + 16);
      b1.h[0] = *(const v8bf*)(kptr1 + ks);
      b1.h[1] = *(const v8bf*)(kptr1 + ks + 16);
      S0 = wmma_bf16(aQ[kc], b0, S0);
      S1 = wmma_bf16(aQ[kc], b1, S1);
    }
    // ---- scale, causal mask, online softmax ----
    int sc0 = s0 + lane15, sc1 = sc0 + 16;
    for (int j = 0; j < 8; ++j) {
      int q = q0 + j + hi * 8;
      float x0 = (sc0 <= q) ? S0[j] * scale : -3.0e38f;
      float x1 = (sc1 <= q) ? S1[j] * scale : -3.0e38f;
      float rm = fmaxf(x0, x1);
      for (int m = 1; m < 16; m <<= 1) rm = fmaxf(rm, __shfl_xor(rm, m, 32));
      float nm = fmaxf(mrow[j], rm);
      float fj = __expf(mrow[j] - nm);
      mrow[j] = nm; f[j] = fj;
      float p0 = __expf(x0 - nm);
      float p1 = __expf(x1 - nm);
      float rs = p0 + p1;
      for (int m = 1; m < 16; m <<= 1) rs += __shfl_xor(rs, m, 32);
      lrow[j] = lrow[j] * fj + rs;
      // stage P (bf16) to LDS in plain 16x32 row-major
      sP[wv][(j + hi * 8) * 32 + lane15]      = (__bf16)p0;
      sP[wv][(j + hi * 8) * 32 + 16 + lane15] = (__bf16)p1;
    }
    // in-wave DS RAW: DS ops are in-order per wave; wait + compiler fence
    asm volatile("s_wait_dscnt 0" ::: "memory");
    // re-read P as an A fragment (16x32)
    V16 pf;
    {
      const __bf16* pr = &sP[wv][lane15 * 32 + hi * 8];
      pf.h[0] = *(const v8bf*)(pr);
      pf.h[1] = *(const v8bf*)(pr + 16);
    }
    // ---- O = O*f + P*V ----
    const __bf16* vbase = Vt + (size_t)kh * HDn * Tn + s0 + hi * 8;
    for (int nt = 0; nt < 8; ++nt) {
      int d = nt * 16 + lane15;
      V16 vf;
      const __bf16* vp = vbase + (size_t)d * Tn;
      vf.h[0] = *(const v8bf*)(vp);
      vf.h[1] = *(const v8bf*)(vp + 16);
      v8f o = O[nt];
      for (int j = 0; j < 8; ++j) o[j] *= f[j];
      O[nt] = wmma_bf16(pf, vf, o);
    }
  }
  // ---- normalize and emit Y (bf16, [T][H*HD]) ----
  float inv[8];
  for (int j = 0; j < 8; ++j) inv[j] = 1.0f / lrow[j];
  for (int nt = 0; nt < 8; ++nt)
    for (int j = 0; j < 8; ++j) {
      int q = q0 + j + hi * 8;
      Yb[(size_t)q * Cn + h * HDn + nt * 16 + lane15] = (__bf16)(O[nt][j] * inv[j]);
    }
}

// ---------------------------------------------------------------------------
extern "C" void kernel_launch(void* const* d_in, const int* in_sizes, int n_in,
                              void* d_out, int out_size, void* d_ws, size_t ws_size,
                              hipStream_t stream) {
  const float* x   = (const float*)d_in[0];
  const float* Wq  = (const float*)d_in[1];
  const float* Wk  = (const float*)d_in[2];
  const float* Wv  = (const float*)d_in[3];
  const float* Wo  = (const float*)d_in[4];
  const int*   pid = (const int*)d_in[5];
  float* out = (float*)d_out;

  char* ws = (char*)d_ws;
  size_t off = 0;
  auto alloc = [&](size_t bytes) { char* p = ws + off; off += (bytes + 255) & ~size_t(255); return p; };

  __bf16* xb   = (__bf16*)alloc((size_t)Tn * Cn * 2);
  __bf16* Wqt  = (__bf16*)alloc((size_t)Cn * Cn * 2);          // [2048][2048]
  __bf16* Wkt  = (__bf16*)alloc((size_t)(KHn * HDn) * Cn * 2); // [512][2048]
  __bf16* Wvt  = (__bf16*)alloc((size_t)(KHn * HDn) * Cn * 2);
  __bf16* Wot  = (__bf16*)alloc((size_t)Cn * Cn * 2);
  float*  Qraw = (float*)alloc((size_t)Tn * Hn * HDn * 4);
  float*  Kraw = (float*)alloc((size_t)Tn * KHn * HDn * 4);
  float*  Vraw = (float*)alloc((size_t)Tn * KHn * HDn * 4);
  __bf16* Qb   = (__bf16*)alloc((size_t)Hn * Tn * HDn * 2);
  __bf16* Kb   = (__bf16*)alloc((size_t)KHn * Tn * HDn * 2);
  __bf16* Vt   = (__bf16*)alloc((size_t)KHn * HDn * Tn * 2);
  __bf16* Yb   = (__bf16*)alloc((size_t)Tn * Cn * 2);

  // 1) x -> bf16
  {
    size_t n = (size_t)Tn * Cn;
    cvt_bf16_kernel<<<(unsigned)((n + 255) / 256), 256, 0, stream>>>(x, xb, n);
  }
  // 2) weight transposes (fp32 [K][N] -> bf16 [N][K])
  {
    size_t n = (size_t)Cn * Cn;
    transpose_bf16_kernel<<<(unsigned)((n + 255) / 256), 256, 0, stream>>>(Wq, Wqt, Cn, Cn);
    transpose_bf16_kernel<<<(unsigned)((n + 255) / 256), 256, 0, stream>>>(Wo, Wot, Cn, Cn);
    size_t nk = (size_t)Cn * (KHn * HDn);
    transpose_bf16_kernel<<<(unsigned)((nk + 255) / 256), 256, 0, stream>>>(Wk, Wkt, Cn, KHn * HDn);
    transpose_bf16_kernel<<<(unsigned)((nk + 255) / 256), 256, 0, stream>>>(Wv, Wvt, Cn, KHn * HDn);
  }
  // 3) projections (TDM-staged bf16 WMMA GEMMs, f32 out)
  {
    int blocks_q = (Tn / 128) * (Cn / 128);            // 256
    gemm_bf16_tdm_kernel<<<blocks_q, 512, 0, stream>>>(xb, Wqt, Qraw, Tn, Cn, Cn);
    int blocks_k = (Tn / 128) * ((KHn * HDn) / 128);   // 64
    gemm_bf16_tdm_kernel<<<blocks_k, 512, 0, stream>>>(xb, Wkt, Kraw, Tn, KHn * HDn, Cn);
    gemm_bf16_tdm_kernel<<<blocks_k, 512, 0, stream>>>(xb, Wvt, Vraw, Tn, KHn * HDn, Cn);
  }
  // 4) RoPE + layout for attention
  rope_kernel<<<dim3(Tn, Hn + KHn), 128, 0, stream>>>(Qraw, Kraw, pid, Qb, Kb);
  prep_v_kernel<<<dim3(Tn / 256, KHn * HDn), 256, 0, stream>>>(Vraw, Vt);
  // 5) flash attention (WMMA QK^T and PV)
  {
    int waves = Hn * (Tn / 16);                        // 2048 waves, 4 per block
    attention_kernel<<<waves / 4, 128, 0, stream>>>(Qb, Kb, Vt, Yb);
  }
  // 6) output projection -> f32 d_out
  {
    int blocks_o = (Tn / 128) * (Cn / 128);
    gemm_bf16_tdm_kernel<<<blocks_o, 512, 0, stream>>>(Yb, Wot, out, Tn, Cn, Cn);
  }
}
